// TemporalGNN_10101763080488
// MI455X (gfx1250) — matrix operand order repaired and checked by reading, not compiled, not run on hardware
//
#include <hip/hip_runtime.h>
#include <hip/hip_bf16.h>
#include <stdint.h>

#define B_  4
#define S_  32
#define N_  1024
#define F_  64
#define H_  128
#define OUT_ 32
#define E_  16384
#define BN_ (B_ * N_)      // 4096
#define G4_ (4 * H_)       // 512 gate columns

typedef __bf16 bf16_t;
typedef __attribute__((ext_vector_type(16))) bf16_t v16bf;
typedef __attribute__((ext_vector_type(8)))  float  v8f;

union ABFrag { unsigned int u[8]; v16bf v; };
union CFrag  { float f[8]; v8f v; };

// ---------- scalar helpers ----------
__device__ __forceinline__ unsigned short f2bf(float f) {
    unsigned int u = __float_as_uint(f);
    unsigned int r = (u + 0x7FFFu + ((u >> 16) & 1u)) >> 16;  // RNE
    return (unsigned short)r;
}
__device__ __forceinline__ unsigned int pack2bf(float lo, float hi) {
    return (unsigned int)f2bf(lo) | ((unsigned int)f2bf(hi) << 16);
}
__device__ __forceinline__ float bf2f(unsigned short u) {
    return __uint_as_float(((unsigned int)u) << 16);
}
__device__ __forceinline__ float sigmoidf_(float x) { return 1.0f / (1.0f + __expf(-x)); }

// ---------- tiny utility kernels ----------
__global__ void k_zero_u32(unsigned int* p, int n) {
    int i = blockIdx.x * blockDim.x + threadIdx.x;
    if (i < n) p[i] = 0u;
}

__global__ void k_deg(const int* ei, const float* ew, float* deg) {
    int e = blockIdx.x * blockDim.x + threadIdx.x;
    if (e < E_) unsafeAtomicAdd(&deg[ei[E_ + e]], ew[e]);   // dst = ei[1][e]
}

__global__ void k_dinv(float* deg) {
    int n = blockIdx.x * blockDim.x + threadIdx.x;
    if (n < N_) deg[n] = rsqrtf(deg[n] + 1.0f);             // in-place dinv
}

__global__ void k_norm(const int* ei, const float* ew, const float* dinv, float* norm) {
    int e = blockIdx.x * blockDim.x + threadIdx.x;
    if (e < E_) norm[e] = dinv[ei[e]] * ew[e] * dinv[ei[E_ + e]];
}

// agg = dinv^2 * x   (self-loop term; also serves as the accumulator init)
__global__ void k_selfloop(const float* __restrict__ x, const float* __restrict__ dinv,
                           float* __restrict__ agg) {
    long long i4 = (long long)blockIdx.x * blockDim.x + threadIdx.x;   // float4 index
    long long total = (long long)B_ * S_ * N_ * F_ / 4;
    if (i4 >= total) return;
    long long base = i4 * 4;
    int n = (int)((base / F_) % N_);
    float d = dinv[n];
    float s = d * d;
    const float4 v = ((const float4*)x)[i4];
    float4 o; o.x = v.x * s; o.y = v.y * s; o.z = v.z * s; o.w = v.w * s;
    ((float4*)agg)[i4] = o;
}

// one thread per (edge, b, s): agg[b,s,dst,:] += norm[e] * x[b,s,src,:]
__global__ void k_edge_agg(const float* __restrict__ x, const int* __restrict__ ei,
                           const float* __restrict__ norm, float* __restrict__ agg) {
    long long idx = (long long)blockIdx.x * blockDim.x + threadIdx.x;
    if (idx >= (long long)E_ * B_ * S_) return;
    int e  = (int)(idx / (B_ * S_));
    int bs = (int)(idx % (B_ * S_));
    int src = ei[e], dst = ei[E_ + e];
    float w = norm[e];
    const float4* xr = (const float4*)(x + ((long long)bs * N_ + src) * F_);
    float* ar = agg + ((long long)bs * N_ + dst) * F_;
#pragma unroll
    for (int f4 = 0; f4 < F_ / 4; ++f4) {
        float4 v = xr[f4];
        unsafeAtomicAdd(&ar[f4 * 4 + 0], w * v.x);
        unsafeAtomicAdd(&ar[f4 * 4 + 1], w * v.y);
        unsafeAtomicAdd(&ar[f4 * 4 + 2], w * v.z);
        unsafeAtomicAdd(&ar[f4 * 4 + 3], w * v.w);
    }
}

// Prepack weights into bf16 WMMA-B fragment-major layout + bias sum.
// pk_lstm: B = [W_ih^T ; W_hh^T], K=256, Ncols=512 -> 32 ct * 8 kk * 32 lanes * 8 dwords
// pk_g:    B = W_gcn,            K=64,  Ncols=128 ->  8 ct * 2 kk * 32 lanes * 8 dwords
__global__ void k_prep(const float* __restrict__ W_ih, const float* __restrict__ W_hh,
                       const float* __restrict__ W_gcn,
                       const float* __restrict__ b_ih, const float* __restrict__ b_hh,
                       unsigned int* __restrict__ pk_lstm, unsigned int* __restrict__ pk_g,
                       float* __restrict__ bias_sum) {
    int idx = blockIdx.x * blockDim.x + threadIdx.x;
    if (idx < 65536) {                       // LSTM weight pack
        int r    = idx & 7;
        int lane = (idx >> 3) & 31;
        int t    = idx >> 8;                 // ct*8 + kk
        int kk   = t & 7;
        int ct   = t >> 3;
        int col, k0;
        if (lane < 16) { col = ct * 16 + lane;        k0 = kk * 32 + 2 * r + ((r >= 4) ? 8 : 0); }
        else           { col = ct * 16 + (lane - 16); k0 = kk * 32 + 8 + 2 * r + ((r >= 4) ? 8 : 0); }
        float lo = (k0     < H_) ? W_ih[col * H_ + k0]       : W_hh[col * H_ + (k0 - H_)];
        float hi = (k0 + 1 < H_) ? W_ih[col * H_ + (k0 + 1)] : W_hh[col * H_ + (k0 + 1 - H_)];
        pk_lstm[idx] = pack2bf(lo, hi);
    } else if (idx < 65536 + 4096) {         // GCN weight pack
        int i2   = idx - 65536;
        int r    = i2 & 7;
        int lane = (i2 >> 3) & 31;
        int t    = i2 >> 8;                  // ct*2 + kk
        int kk   = t & 1;
        int ct   = t >> 1;
        int col = ct * 16 + (lane & 15);                       // h column
        int k0  = kk * 32 + ((lane < 16) ? 0 : 8) + 2 * r + ((r >= 4) ? 8 : 0);    // f row
        pk_g[i2] = pack2bf(W_gcn[k0 * H_ + col], W_gcn[(k0 + 1) * H_ + col]);
    } else if (idx < 65536 + 4096 + 512) {
        int j = idx - 65536 - 4096;
        bias_sum[j] = b_ih[j] + b_hh[j];
    }
}

// ---------- fragment builders ----------
// A (16x32 bf16) per ISA table: lanes 0-15 rows, v0..3 = K{0..7}, v4..7 = K{16..23};
// lanes 16-31 same rows, K{8..15} and K{24..31}.
__device__ __forceinline__ void load_a_frag(ABFrag& a, const unsigned int* ldsA,
                                            int rowStrideDw, int kk, int lane) {
    int row = lane & 15;
    int off = (lane >= 16) ? 4 : 0;
    const unsigned int* p = ldsA + row * rowStrideDw + kk * 16 + off;
#pragma unroll
    for (int i = 0; i < 4; ++i) { a.u[i] = p[i]; a.u[4 + i] = p[8 + i]; }
}
__device__ __forceinline__ void load_b_frag(ABFrag& b, const unsigned int* pk,
                                            int tileK, int lane) {
    const uint4* q = (const uint4*)(pk + ((long long)tileK * 32 + lane) * 8);
    uint4 q0 = q[0], q1 = q[1];
    b.u[0] = q0.x; b.u[1] = q0.y; b.u[2] = q0.z; b.u[3] = q0.w;
    b.u[4] = q1.x; b.u[5] = q1.y; b.u[6] = q1.z; b.u[7] = q1.w;
}

// ---------- GCN linear: seq[s, b*N+n, h] = agg[b,s,n,:] @ W_gcn ----------
// grid.x = (B*S) * (N/16); block = 256 (8 waves, one 16-col tile each)
__global__ void k_gcn_gemm(const float* __restrict__ agg, const unsigned int* __restrict__ pk_g,
                           unsigned short* __restrict__ seq) {
    __shared__ unsigned int ldsA[16 * 32];         // 16 rows x 64 bf16
    int bs = blockIdx.x >> 6;                      // b*S + s
    int nt = blockIdx.x & 63;
    int n0 = nt * 16;
    int b  = bs >> 5, s = bs & 31;
    const float* src = agg + ((long long)bs * N_ + n0) * F_;

    for (int d = threadIdx.x; d < 512; d += 256) { // pack fp32 -> bf16 pairs into LDS
        int row = d >> 5, kd2 = d & 31;
        float lo = src[row * F_ + kd2 * 2];
        float hi = src[row * F_ + kd2 * 2 + 1];
        ldsA[row * 32 + kd2] = pack2bf(lo, hi);
    }
    __syncthreads();

    int wave = threadIdx.x >> 5;
    int lane = threadIdx.x & 31;
    int ct = wave;                                 // column tile 0..7 -> h cols 16*ct..
    CFrag c;
#pragma unroll
    for (int r = 0; r < 8; ++r) c.f[r] = 0.0f;     // b_gcn == 0
#pragma unroll
    for (int kk = 0; kk < 2; ++kk) {
        ABFrag a, bfr;
        load_a_frag(a, ldsA, 32, kk, lane);
        load_b_frag(bfr, pk_g, ct * 2 + kk, lane);
        c.v = __builtin_amdgcn_wmma_f32_16x16x32_bf16(false, a.v, false, bfr.v,
                                                      (short)0, c.v, false, false);
    }
    // D layout: vgpr r -> row (lane<16 ? r : r+8), col = ct*16 + (lane&15)
    int colBase = ct * 16 + (lane & 15);
    int rowAdd  = (lane < 16) ? 0 : 8;
#pragma unroll
    for (int r = 0; r < 8; ++r) {
        int n = n0 + rowAdd + r;
        long long oi = (((long long)s * BN_) + (long long)b * N_ + n) * H_ + colBase;
        seq[oi] = f2bf(c.f[r]);
    }
}

// ---------- LSTM step ----------
// grid.x = BN/16 = 256 blocks; block 256 threads (8 waves x 4 column tiles of 16).
// A strip [x_t | h_prev] staged via GLOBAL_LOAD_ASYNC_TO_LDS_B128 (ASYNCcnt path).
__global__ void k_lstm_step(int s, const unsigned short* __restrict__ seq,
                            const unsigned short* __restrict__ hin,
                            unsigned short* __restrict__ hout,
                            float* __restrict__ cbuf,
                            const unsigned int* __restrict__ pk_lstm,
                            const float* __restrict__ bias_sum) {
    __shared__ unsigned int ldsA[16 * 128];        // 16 rows x 256 bf16 = [x_t | h_prev]
    __shared__ float gates[16 * G4_];              // 16 x 512 fp32

    int bn0 = blockIdx.x * 16;
    const unsigned int* xsrc = (const unsigned int*)(seq + ((long long)s * BN_ + bn0) * H_);
    const unsigned int* hsrc = (const unsigned int*)(hin + (long long)bn0 * H_);

    // 512 x 16B chunks, 2 per thread: rows of 32 chunks (16 from x, 16 from h)
#pragma unroll
    for (int i = 0; i < 2; ++i) {
        int cidx = threadIdx.x + i * 256;
        int row  = cidx >> 5;
        int q    = cidx & 31;
        const unsigned int* g = (q < 16) ? (xsrc + row * 64 + q * 4)
                                         : (hsrc + row * 64 + (q - 16) * 4);
        // low 32 bits of the generic pointer == LDS byte offset (flat-aperture mapping)
        unsigned lds_off = (unsigned)(unsigned long long)
                           (&ldsA[row * 128 + ((q < 16) ? q * 4 : 64 + (q - 16) * 4)]);
        asm volatile("global_load_async_to_lds_b128 %0, %1, off"
                     :: "v"(lds_off), "v"(g) : "memory");
    }
    asm volatile("s_wait_asynccnt 0x0" ::: "memory");
    __syncthreads();

    int wave = threadIdx.x >> 5;
    int lane = threadIdx.x & 31;
    int rowAdd = (lane < 16) ? 0 : 8;

    CFrag acc[4];
#pragma unroll
    for (int tl = 0; tl < 4; ++tl) {
        float bias = bias_sum[(wave * 4 + tl) * 16 + (lane & 15)];
#pragma unroll
        for (int r = 0; r < 8; ++r) acc[tl].f[r] = bias;
    }
    // K = 256 (x then h); A fragment loaded once per kk, reused across 4 column tiles
#pragma unroll
    for (int kk = 0; kk < 8; ++kk) {
        ABFrag a;
        load_a_frag(a, ldsA, 128, kk, lane);
#pragma unroll
        for (int tl = 0; tl < 4; ++tl) {
            int ct = wave * 4 + tl;
            ABFrag bfr;
            load_b_frag(bfr, pk_lstm, ct * 8 + kk, lane);
            acc[tl].v = __builtin_amdgcn_wmma_f32_16x16x32_bf16(false, a.v, false, bfr.v,
                                                                (short)0, acc[tl].v,
                                                                false, false);
        }
    }
#pragma unroll
    for (int tl = 0; tl < 4; ++tl) {
        int colBase = (wave * 4 + tl) * 16 + (lane & 15);
#pragma unroll
        for (int r = 0; r < 8; ++r)
            gates[(rowAdd + r) * G4_ + colBase] = acc[tl].f[r];
    }
    __syncthreads();

    // pointwise: 16 rows x 128 h = 2048 items, 8 consecutive per thread
    int base = threadIdx.x * 8;
    int row = base >> 7;
    int j0  = base & 127;
    const float* g = gates + row * G4_;
    long long gi = (long long)(bn0 + row) * H_ + j0;
#pragma unroll
    for (int k = 0; k < 8; ++k) {
        int j = j0 + k;
        float ig = sigmoidf_(g[j]);
        float fg = sigmoidf_(g[H_ + j]);
        float gg = tanhf(g[2 * H_ + j]);
        float og = sigmoidf_(g[3 * H_ + j]);
        float cn = fg * cbuf[gi + k] + ig * gg;
        cbuf[gi + k] = cn;
        hout[gi + k] = f2bf(og * tanhf(cn));
    }
}

// ---------- FC: out[b,n,o] = h_last[bn,:] . W_fc[o,:] + b_fc[o] ----------
__global__ void k_fc(const unsigned short* __restrict__ hlast,
                     const float* __restrict__ W_fc, const float* __restrict__ b_fc,
                     float* __restrict__ out) {
    int idx = blockIdx.x * blockDim.x + threadIdx.x;
    if (idx >= BN_ * OUT_) return;
    int o  = idx & (OUT_ - 1);
    int bn = idx >> 5;
    const unsigned short* h = hlast + (long long)bn * H_;
    const float* w = W_fc + o * H_;
    float acc = b_fc[o];
#pragma unroll 8
    for (int k = 0; k < H_; ++k) acc += bf2f(h[k]) * w[k];
    out[(long long)bn * OUT_ + o] = acc;
}

// ---------- host side ----------
extern "C" void kernel_launch(void* const* d_in, const int* in_sizes, int n_in,
                              void* d_out, int out_size, void* d_ws, size_t ws_size,
                              hipStream_t stream) {
    const float* x        = (const float*)d_in[0];
    const int*   ei       = (const int*)d_in[1];
    const float* ew       = (const float*)d_in[2];
    const float* W_gcn    = (const float*)d_in[3];
    // d_in[4] = b_gcn (zeros, unused)
    const float* W_ih     = (const float*)d_in[5];
    const float* W_hh     = (const float*)d_in[6];
    const float* b_ih     = (const float*)d_in[7];
    const float* b_hh     = (const float*)d_in[8];
    const float* W_fc     = (const float*)d_in[9];
    const float* b_fc     = (const float*)d_in[10];
    float* out = (float*)d_out;

    char* ws = (char*)d_ws;
    size_t off = 0;
    auto alloc = [&](size_t bytes) { char* p = ws + off; off = (off + bytes + 255) & ~(size_t)255; return p; };
    float*          deg      = (float*)alloc((size_t)N_ * 4);             // becomes dinv
    float*          norm     = (float*)alloc((size_t)E_ * 4);
    float*          agg      = (float*)alloc((size_t)B_ * S_ * N_ * F_ * 4);
    unsigned short* seq      = (unsigned short*)alloc((size_t)S_ * BN_ * H_ * 2);
    unsigned short* hbuf0    = (unsigned short*)alloc((size_t)BN_ * H_ * 2);
    unsigned short* hbuf1    = (unsigned short*)alloc((size_t)BN_ * H_ * 2);
    float*          cbuf     = (float*)alloc((size_t)BN_ * H_ * 4);
    unsigned int*   pk_lstm  = (unsigned int*)alloc(65536u * 4);
    unsigned int*   pk_g     = (unsigned int*)alloc(4096u * 4);
    float*          bias_sum = (float*)alloc(512u * 4);
    (void)ws_size; (void)in_sizes; (void)n_in; (void)out_size;

    const int T = 256;
    // init
    k_zero_u32<<<(N_ + T - 1) / T, T, 0, stream>>>((unsigned int*)deg, N_);
    k_zero_u32<<<(BN_ * H_ / 2 + T - 1) / T, T, 0, stream>>>((unsigned int*)hbuf0, BN_ * H_ / 2);
    k_zero_u32<<<(BN_ * H_ + T - 1) / T, T, 0, stream>>>((unsigned int*)cbuf, BN_ * H_);
    // graph norm
    k_deg<<<E_ / T, T, 0, stream>>>(ei, ew, deg);
    k_dinv<<<N_ / T, T, 0, stream>>>(deg);
    k_norm<<<E_ / T, T, 0, stream>>>(ei, ew, deg, norm);
    // weight prep
    k_prep<<<(65536 + 4096 + 512 + T - 1) / T, T, 0, stream>>>(W_ih, W_hh, W_gcn, b_ih, b_hh,
                                                               pk_lstm, pk_g, bias_sum);
    // aggregation
    long long nSelf = (long long)B_ * S_ * N_ * F_ / 4;
    k_selfloop<<<(int)((nSelf + T - 1) / T), T, 0, stream>>>(x, deg, agg);
    long long nEdge = (long long)E_ * B_ * S_;
    k_edge_agg<<<(int)((nEdge + T - 1) / T), T, 0, stream>>>(x, ei, norm, agg);
    // GCN linear -> seq (bf16, [S, B*N, H])
    k_gcn_gemm<<<B_ * S_ * (N_ / 16), T, 0, stream>>>(agg, pk_g, seq);
    // LSTM: 32 sequential step launches, ping-pong h buffers
    for (int s = 0; s < S_; ++s) {
        const unsigned short* hin = (s & 1) ? hbuf1 : hbuf0;
        unsigned short* hout      = (s & 1) ? hbuf0 : hbuf1;
        k_lstm_step<<<BN_ / 16, T, 0, stream>>>(s, seq, hin, hout, cbuf, pk_lstm, bias_sum);
    }
    // final h is in hbuf0 (S even)
    k_fc<<<(BN_ * OUT_ + T - 1) / T, T, 0, stream>>>(hbuf0, W_fc, b_fc, out);
}
